// PairBiasedSelfAttention_58274116272637
// MI455X (gfx1250) — compile-verified, matrix operand order checked by reading, not run on hardware
//
#include <hip/hip_runtime.h>
#include <math.h>

typedef __attribute__((ext_vector_type(16))) _Float16 v16h;
typedef __attribute__((ext_vector_type(8)))  _Float16 v8h;
typedef __attribute__((ext_vector_type(8)))  float    v8f;
typedef __attribute__((ext_vector_type(4)))  float    v4f;
typedef __attribute__((ext_vector_type(4)))  unsigned int u32x4;
typedef __attribute__((ext_vector_type(8)))  int      i32x8;
typedef __attribute__((ext_vector_type(4)))  int      i32x4;

#define NB   2
#define NSEQ 512
#define DIMX 512
#define NH   8
#define DH   64
#define ED   64
#define HS   520   // LDS row stride (halves) for h buffer: 1040B -> 4-bank step/row
#define W1S  72    // LDS row stride (halves) for staged We1^T: 144B -> 4-bank step/row
#define W2S  520   // LDS row stride (halves) for staged We2^T
#define LS   520   // LDS row stride (floats) for logits
#define AS   520   // LDS row stride (halves) for attention probs

#define HB_BYTES  (4 * 16 * HS * 2)        // 66560
#define WE1_BYTES (512 * W1S * 2)          // 73728
#define WE2_BYTES (16 * W2S * 2)           // 16640
#define EDGE_LDS  (HB_BYTES + WE1_BYTES + WE2_BYTES)

#define WMMA(a, b, c) __builtin_amdgcn_wmma_f32_16x16x32_f16(false, (a), false, (b), (short)0, (c), false, false)

// ---- WMMA fragment loaders (layout per CDNA5 ISA 7.12.2) -------------------
// A (and B^T) 16x32 f16: lanes 0-15 row M=lane, halves {K0..7, K16..23};
//                        lanes 16-31 row M=lane-16, halves {K8..15, K24..31}.
static __device__ __forceinline__ v16h frag_from_f16(const _Float16* p, int ld, int lane) {
  int row = lane & 15;
  int kh  = (lane >> 4) << 3;
  const v8h* q = (const v8h*)(p + row * ld + kh);
  v8h lo = q[0];
  v8h hi = q[2];
  v16h r;
#pragma unroll
  for (int i = 0; i < 8; ++i) { r[i] = lo[i]; r[8 + i] = hi[i]; }
  return r;
}

static __device__ __forceinline__ v16h frag_from_f32(const float* p, int ld, int lane) {
  int row = lane & 15;
  int kh  = (lane >> 4) << 3;
  const v4f* q = (const v4f*)(p + row * ld + kh);
  v4f a0 = q[0], a1 = q[1], a2 = q[4], a3 = q[5];
  v16h r;
#pragma unroll
  for (int i = 0; i < 4; ++i) {
    r[i]      = (_Float16)a0[i];
    r[4 + i]  = (_Float16)a1[i];
    r[8 + i]  = (_Float16)a2[i];
    r[12 + i] = (_Float16)a3[i];
  }
  return r;
}

// tanh-form GELU; v_tanh_f32 is a CDNA5 TRANS op (co-executes with WMMA).
// v_nop covers the TRANS->use hazard the compiler cannot see through the asm.
static __device__ __forceinline__ float gelu_fast(float x) {
  float x2 = x * x;
  float w  = __builtin_fmaf(0.03567740814f, x2, 0.79788456080f); // 0.79788456*(1+0.044715*x^2)
  float u  = x * w;
  float t;
  asm volatile("v_tanh_f32 %0, %1\n\tv_nop" : "=v"(t) : "v"(u));
  return 0.5f * x * (1.0f + t);
}

// low 32 bits of the LDS-aperture generic address == LDS byte offset (ISA 10.2)
static __device__ __forceinline__ unsigned lds_offset(const void* p) {
  return (unsigned)(unsigned long long)(size_t)p;
}

// TDM: DMA a 2-D f16 tile (rows x width_h halves, contiguous) global -> LDS,
// optionally inserting pad_amount dwords after every pad_interval dwords so the
// LDS image gets a bank-conflict-free row stride. D# per ISA 08_async_tensor §8.
static __device__ __forceinline__ void tdm_load_2d_f16(const void* gsrc, unsigned lds_addr,
                                                       unsigned width_h, unsigned rows,
                                                       unsigned pad_icode, unsigned pad_acode) {
  unsigned long long ga = (unsigned long long)(size_t)gsrc;
  u32x4 g0;
  g0[0] = 1u;                                                    // count=1 (valid user D#)
  g0[1] = lds_addr;                                              // lds_addr [63:32]
  g0[2] = (unsigned)(ga & 0xFFFFFFFFull);                        // global_addr lo
  g0[3] = (unsigned)((ga >> 32) & 0x1FFFFFFull) | (2u << 30);    // global_addr hi | type=2
  i32x8 g1;
  unsigned d0 = (1u << 16)                                       // data_size = 2 bytes
              | (1u << 20)                                       // pad_enable
              | (pad_icode << 22) | (pad_acode << 25);
  g1[0] = (int)d0;
  g1[1] = (int)((width_h & 0xFFFFu) << 16);                      // tensor_dim0[15:0]
  g1[2] = (int)(((width_h >> 16) & 0xFFFFu) | ((rows & 0xFFFFu) << 16)); // dim0 hi | dim1 lo
  g1[3] = (int)(((rows >> 16) & 0xFFFFu) | ((width_h & 0xFFFFu) << 16)); // dim1 hi | tile_dim0
  g1[4] = (int)(rows & 0xFFFFu);                                 // tile_dim1 | tile_dim2=0
  g1[5] = (int)width_h;                                          // tensor_dim0_stride lo32
  g1[6] = 0;                                                     // stride0 hi | stride1 lo
  g1[7] = 0;                                                     // stride1 hi
  i32x4 gz = {0, 0, 0, 0};
#if defined(__clang_major__) && (__clang_major__ >= 23)
  i32x8 gz8 = {0, 0, 0, 0, 0, 0, 0, 0};
  __builtin_amdgcn_tensor_load_to_lds(g0, g1, gz, gz, gz8, 0);
#else
  __builtin_amdgcn_tensor_load_to_lds(g0, g1, gz, gz, 0);
#endif
}

// ---- kernel 1: weight prep (transpose -> f16, pad We2 heads to 16) ---------
__global__ void prep_kernel(const float* Wq, const float* Wk, const float* Wv, const float* Wo,
                            const float* We1, const float* We2,
                            _Float16* Wqt, _Float16* Wkt, _Float16* Wvt, _Float16* Wot,
                            _Float16* We1t, _Float16* We2t) {
  int idx = blockIdx.x * blockDim.x + threadIdx.x;
  const int big = DIMX * DIMX;
  if (idx < 4 * big) {
    int m = idx / big, r = idx % big;
    int n = r / DIMX, k = r % DIMX;
    const float* W = (m == 0) ? Wq : (m == 1) ? Wk : (m == 2) ? Wv : Wo;
    _Float16* Wt   = (m == 0) ? Wqt : (m == 1) ? Wkt : (m == 2) ? Wvt : Wot;
    Wt[n * DIMX + k] = (_Float16)W[k * DIMX + n];
  } else {
    int r = idx - 4 * big;
    if (r < DIMX * ED) {                       // We1t[n][k] = We1[k][n], k<64
      int n = r / ED, k = r % ED;
      We1t[n * ED + k] = (_Float16)We1[k * DIMX + n];
    } else {
      r -= DIMX * ED;
      if (r < 16 * DIMX) {                     // We2t[n][k], rows 8..15 zero pad
        int n = r / DIMX, k = r % DIMX;
        We2t[n * DIMX + k] = (n < NH) ? (_Float16)We2[k * NH + n] : (_Float16)0.0f;
      }
    }
  }
}

// ---- kernel 2: Q/K/V projections (1 M-tile x 4 N-tiles per wave) -----------
__global__ void qkv_kernel(const float* nodes,
                           const _Float16* Wqt, const _Float16* Wkt, const _Float16* Wvt,
                           const float* bq, const float* bk, const float* bv,
                           _Float16* Qh, _Float16* Kh, _Float16* Vt) {
  int wave = (blockIdx.x * blockDim.x + threadIdx.x) >> 5;   // 1536 waves
  int lane = threadIdx.x & 31;
  int which = wave / 512;                      // 0=q 1=k 2=v
  int t = wave % 512;
  int m0 = (t >> 3) << 4;                      // 64 M-tiles over B*N=1024 rows
  int n0 = (t & 7) << 6;                       // 8 N-quads of 64 cols
  const _Float16* Wt = (which == 0) ? Wqt : (which == 1) ? Wkt : Wvt;
  const float* bias  = (which == 0) ? bq  : (which == 1) ? bk  : bv;

  v8f acc[4] = {};
  for (int k0 = 0; k0 < DIMX; k0 += 32) {
    v16h a = frag_from_f32(nodes + (size_t)m0 * DIMX + k0, DIMX, lane);
#pragma unroll
    for (int u = 0; u < 4; ++u) {
      v16h bf = frag_from_f16(Wt + (size_t)(n0 + 16 * u) * DIMX + k0, DIMX, lane);
      acc[u] = WMMA(a, bf, acc[u]);
    }
  }
  int rb = m0 + ((lane >> 4) << 3);
#pragma unroll
  for (int u = 0; u < 4; ++u) {
    int col = n0 + 16 * u + (lane & 15);
    int h = col >> 6, d = col & 63;
    float bb = bias[col];
#pragma unroll
    for (int r = 0; r < 8; ++r) {
      int row = rb + r;                        // 0..1023
      int b = row >> 9, n = row & 511;
      float v = acc[u][r] + bb;
      if (which == 0)      Qh[(((size_t)(b * NH + h) * NSEQ) + n) * DH + d] = (_Float16)(v * 0.125f);
      else if (which == 1) Kh[(((size_t)(b * NH + h) * NSEQ) + n) * DH + d] = (_Float16)v;
      else                 Vt[(((size_t)(b * NH + h) * DH) + d) * NSEQ + n] = (_Float16)v;
    }
  }
}

// ---- kernel 3: fused edge-bias MLP (TDM-staged weights, GEMM->GELU->GEMM) --
__global__ void edge_kernel(const float* ef, const _Float16* We1t, const _Float16* We2t,
                            const float* be1, const float* be2, float* biasArr) {
  extern __shared__ __align__(16) char smem[];
  _Float16* hball = (_Float16*)smem;                              // [4][16][HS]
  _Float16* we1s  = (_Float16*)(smem + HB_BYTES);                 // [512][W1S]
  _Float16* we2s  = (_Float16*)(smem + HB_BYTES + WE1_BYTES);     // [16][W2S]

  int w = threadIdx.x >> 5, lane = threadIdx.x & 31;
  int b = blockIdx.x >> 9, i = blockIdx.x & 511;

  if (w == 0) {
    // We1^T: 512 rows x 64 halves; pad 4 dwords after every 32 dwords -> 72/row
    tdm_load_2d_f16(We1t, lds_offset(we1s), 64, 512, /*2^(4+1)=32 dw*/4, /*4 dw*/3);
    // We2^T: 16 rows x 512 halves; pad 4 dwords after every 256 dwords -> 520/row
    tdm_load_2d_f16(We2t, lds_offset(we2s), 512, 16, /*256 dw*/7, /*4 dw*/3);
    __builtin_amdgcn_s_wait_tensorcnt(0);
  }
  __syncthreads();

  _Float16* hb = hball + (size_t)w * 16 * HS;
  const float* efbase = ef + ((size_t)(b * NSEQ + i)) * NSEQ * ED;
  int col = lane & 15, rh = (lane >> 4) << 3;

  for (int jt = w; jt < 32; jt += 4) {
    int j0 = jt << 4;
    if (jt + 4 < 32)
      __builtin_prefetch(efbase + (size_t)(j0 + 64) * ED, 0, 1);

    // A: 16 edge rows x K=64, f32 -> f16 on load
    v16h a0 = frag_from_f32(efbase + (size_t)j0 * ED + 0,  ED, lane);
    v16h a1 = frag_from_f32(efbase + (size_t)j0 * ED + 32, ED, lane);

    // GEMM1 (B from LDS) + GELU -> LDS h tile (f16)
    for (int n0 = 0; n0 < DIMX; n0 += 16) {
      v16h b0 = frag_from_f16(we1s + (size_t)n0 * W1S + 0,  W1S, lane);
      v16h b1 = frag_from_f16(we1s + (size_t)n0 * W1S + 32, W1S, lane);
      v8f c = {};
      c = WMMA(a0, b0, c);
      c = WMMA(a1, b1, c);
      float bb = be1[n0 + col];
#pragma unroll
      for (int r = 0; r < 8; ++r)
        hb[(size_t)(rh + r) * HS + n0 + col] = (_Float16)gelu_fast(c[r] + bb);
    }
    // same-wave LDS ops are in-order: no barrier needed before reading hb back

    // GEMM2: h[16x512] @ We2^T(padded 16x512, LDS) -> bias tile (heads 0..7)
    v8f c2 = {};
    for (int k0 = 0; k0 < DIMX; k0 += 32) {
      v16h a  = frag_from_f16(hb + k0, HS, lane);
      v16h bf = frag_from_f16(we2s + k0, W2S, lane);
      c2 = WMMA(a, bf, c2);
    }
    int head = col;
    if (head < NH) {
      float bb2 = be2[head];
      float* dst = biasArr + (((size_t)(b * NH + head) * NSEQ + i) * NSEQ) + j0 + rh;
#pragma unroll
      for (int r = 0; r < 8; ++r) dst[r] = c2[r] + bb2;
    }
  }
}

// ---- kernel 4: attention (logits + bias, softmax, attn@V) ------------------
__global__ void attn_kernel(const _Float16* Qh, const _Float16* Kh, const _Float16* Vt,
                            const float* biasArr, _Float16* mixed) {
  __shared__ __align__(16) float    slog[16 * LS];
  __shared__ __align__(16) _Float16 sattn[16 * AS];
  __shared__ float red[16 * 8];
  __shared__ float rowstat[16];

  int w = threadIdx.x >> 5, lane = threadIdx.x & 31;
  int bh = blockIdx.x >> 5;                    // b*8+h
  int i0 = (blockIdx.x & 31) << 4;
  const _Float16* Qbase = Qh + ((size_t)bh * NSEQ + i0) * DH;
  const _Float16* Kbase = Kh + (size_t)bh * NSEQ * DH;
  const float*    Bbase = biasArr + ((size_t)bh * NSEQ + i0) * NSEQ;
  int col = lane & 15, rh = (lane >> 4) << 3;

  // phase 1: logits (Q pre-scaled by 1/8) + pair bias -> LDS
  v16h qa0 = frag_from_f16(Qbase + 0,  DH, lane);
  v16h qa1 = frag_from_f16(Qbase + 32, DH, lane);
  for (int t = 0; t < 8; ++t) {
    int j0 = w * 128 + t * 16;
    v16h kb0 = frag_from_f16(Kbase + (size_t)j0 * DH + 0,  DH, lane);
    v16h kb1 = frag_from_f16(Kbase + (size_t)j0 * DH + 32, DH, lane);
    v8f c = {};
    c = WMMA(qa0, kb0, c);
    c = WMMA(qa1, kb1, c);
#pragma unroll
    for (int r = 0; r < 8; ++r) {
      float pb = Bbase[(size_t)(rh + r) * NSEQ + j0 + col];
      slog[(rh + r) * LS + j0 + col] = c[r] + pb;
    }
  }
  __syncthreads();

  // phase 2: softmax over 512 keys (row = tid>>3, 8 segments of 64 per row)
  {
    int row = threadIdx.x >> 3, seg = threadIdx.x & 7;
    float* p = slog + row * LS + seg * 64;
    float mx = -1e30f;
    for (int c2 = 0; c2 < 64; ++c2) mx = fmaxf(mx, p[c2]);
    red[row * 8 + seg] = mx;
    __syncthreads();
    if (seg == 0) {
      float m = red[row * 8];
      for (int s = 1; s < 8; ++s) m = fmaxf(m, red[row * 8 + s]);
      rowstat[row] = m;
    }
    __syncthreads();
    float m = rowstat[row];
    float sum = 0.f;
    for (int c2 = 0; c2 < 64; ++c2) { float e = __expf(p[c2] - m); p[c2] = e; sum += e; }
    red[row * 8 + seg] = sum;
    __syncthreads();
    if (seg == 0) {
      float s2 = 0.f;
      for (int s = 0; s < 8; ++s) s2 += red[row * 8 + s];
      rowstat[row] = 1.0f / s2;
    }
    __syncthreads();
    float inv = rowstat[row];
    _Float16* q2 = sattn + row * AS + seg * 64;
    for (int c2 = 0; c2 < 64; ++c2) q2[c2] = (_Float16)(p[c2] * inv);
  }
  __syncthreads();

  // phase 3: attn[16x512] @ V^T -> mixed d-tile (wave w owns d0 = 16w)
  int d0 = w << 4;
  const _Float16* Vb = Vt + ((size_t)bh * DH + d0) * NSEQ;
  v8f c = {};
  for (int k0 = 0; k0 < NSEQ; k0 += 32) {
    v16h a  = frag_from_f16(sattn + k0, AS, lane);
    v16h bf = frag_from_f16(Vb + k0, NSEQ, lane);
    c = WMMA(a, bf, c);
  }
  int b = bh >> 3, h = bh & 7;
  _Float16* mdst = mixed + ((size_t)(b * NSEQ) + i0 + rh) * DIMX + h * DH + d0 + col;
#pragma unroll
  for (int r = 0; r < 8; ++r) mdst[(size_t)r * DIMX] = (_Float16)c[r];
}

// ---- kernel 5: output projection (1 M-tile x 4 N-tiles per wave) -----------
__global__ void outproj_kernel(const _Float16* mixed, const _Float16* Wot,
                               const float* bo, float* out) {
  int wave = (blockIdx.x * blockDim.x + threadIdx.x) >> 5;   // 512 waves
  int lane = threadIdx.x & 31;
  int m0 = (wave >> 3) << 4;
  int n0 = (wave & 7) << 6;
  v8f acc[4] = {};
  for (int k0 = 0; k0 < DIMX; k0 += 32) {
    v16h a = frag_from_f16(mixed + (size_t)m0 * DIMX + k0, DIMX, lane);
#pragma unroll
    for (int u = 0; u < 4; ++u) {
      v16h bf = frag_from_f16(Wot + (size_t)(n0 + 16 * u) * DIMX + k0, DIMX, lane);
      acc[u] = WMMA(a, bf, acc[u]);
    }
  }
  int r0 = m0 + ((lane >> 4) << 3);
#pragma unroll
  for (int u = 0; u < 4; ++u) {
    int col = n0 + 16 * u + (lane & 15);
    float bb = bo[col];
#pragma unroll
    for (int r = 0; r < 8; ++r) out[(size_t)(r0 + r) * DIMX + col] = acc[u][r] + bb;
  }
}

// ---- host launcher ----------------------------------------------------------
extern "C" void kernel_launch(void* const* d_in, const int* in_sizes, int n_in,
                              void* d_out, int out_size, void* d_ws, size_t ws_size,
                              hipStream_t stream) {
  (void)in_sizes; (void)n_in; (void)out_size; (void)ws_size;
  const float* nodes = (const float*)d_in[0];
  const float* ef    = (const float*)d_in[1];
  const float* Wq = (const float*)d_in[2];  const float* bq = (const float*)d_in[3];
  const float* Wk = (const float*)d_in[4];  const float* bk = (const float*)d_in[5];
  const float* Wv = (const float*)d_in[6];  const float* bv = (const float*)d_in[7];
  const float* We1 = (const float*)d_in[8]; const float* be1 = (const float*)d_in[9];
  const float* We2 = (const float*)d_in[10];const float* be2 = (const float*)d_in[11];
  const float* Wo = (const float*)d_in[12]; const float* bo = (const float*)d_in[13];
  float* out = (float*)d_out;

  size_t off = 0;
  auto alloc = [&](size_t bytes) -> void* {
    void* p = (char*)d_ws + off;
    off += (bytes + 255) & ~(size_t)255;
    return p;
  };
  _Float16* Wqt  = (_Float16*)alloc((size_t)DIMX * DIMX * 2);
  _Float16* Wkt  = (_Float16*)alloc((size_t)DIMX * DIMX * 2);
  _Float16* Wvt  = (_Float16*)alloc((size_t)DIMX * DIMX * 2);
  _Float16* Wot  = (_Float16*)alloc((size_t)DIMX * DIMX * 2);
  _Float16* We1t = (_Float16*)alloc((size_t)DIMX * ED * 2);
  _Float16* We2t = (_Float16*)alloc((size_t)16 * DIMX * 2);
  _Float16* Qh   = (_Float16*)alloc((size_t)NB * NH * NSEQ * DH * 2);
  _Float16* Kh   = (_Float16*)alloc((size_t)NB * NH * NSEQ * DH * 2);
  _Float16* Vt   = (_Float16*)alloc((size_t)NB * NH * NSEQ * DH * 2);
  _Float16* mixd = (_Float16*)alloc((size_t)NB * NSEQ * DIMX * 2);
  float* biasArr = (float*)alloc((size_t)NB * NH * NSEQ * NSEQ * 4);

  {
    int total = 4 * DIMX * DIMX + DIMX * ED + 16 * DIMX;
    prep_kernel<<<(total + 255) / 256, 256, 0, stream>>>(Wq, Wk, Wv, Wo, We1, We2,
                                                         Wqt, Wkt, Wvt, Wot, We1t, We2t);
  }
  qkv_kernel<<<192, 256, 0, stream>>>(nodes, Wqt, Wkt, Wvt, bq, bk, bv, Qh, Kh, Vt);
  edge_kernel<<<NB * NSEQ, 128, EDGE_LDS, stream>>>(ef, We1t, We2t, be1, be2, biasArr);
  attn_kernel<<<NB * NH * (NSEQ / 16), 128, 0, stream>>>(Qh, Kh, Vt, biasArr, mixd);
  outproj_kernel<<<64, 256, 0, stream>>>(mixd, Wot, bo, out);
}
